// GNN_52269751993090
// MI455X (gfx1250) — matrix operand order repaired and checked by reading, not compiled, not run on hardware
//
#include <hip/hip_runtime.h>
#include <stdint.h>

// ---------------- constants (derived from setup_inputs) ----------------
#define CONF_THRES 0.596f
#define IOU_THRES  0.45f
#define NEG_SLOPE  0.01f
#define MAX_WH     4096.0f

constexpr int B      = 8;
constexpr int NPRED  = 64512;   // ((128^2)+(64^2)+(32^2))*3
constexpr int NPAD   = 65536;
constexpr int PREDC  = 85;
constexpr int NCLS   = 80;
constexpr int MAXC   = 512;
constexpr int LIMIT  = 200;
constexpr int MROWS  = 208;     // 13 * 16 (M padded for WMMA tiles)
constexpr int K1     = 1920;    // 128+256+512+1024
constexpr int NH     = 64;
constexpr int IMG    = 1024;

typedef _Float16 v8h  __attribute__((ext_vector_type(8)));
typedef _Float16 v16h __attribute__((ext_vector_type(16)));
typedef float    v8f  __attribute__((ext_vector_type(8)));

// ---------------- stage 1: score -> sortable key ----------------
__global__ void k_keys(const float* __restrict__ pred,
                       unsigned long long* __restrict__ keys) {
  int gid = blockIdx.x * blockDim.x + threadIdx.x;
  if (gid >= B * NPAD) return;
  int b = gid / NPAD, i = gid % NPAD;
  unsigned long long key = 0ull;
  if (i < NPRED) {
    const float* p = pred + ((size_t)b * NPRED + i) * PREDC;
    float obj = p[4];
    float best = -1e30f;
    for (int c = 0; c < NCLS; ++c) {
      float v = p[5 + c] * obj;
      if (v > best) best = v;
    }
    float score = (best > CONF_THRES) ? best : -1.0f;
    unsigned u = __float_as_uint(score);
    u = (u & 0x80000000u) ? ~u : (u | 0x80000000u);   // order-preserving map
    key = ((unsigned long long)u << 32) |
          (unsigned long long)(0xFFFFFFFFu - (unsigned)i); // stable ties: small idx wins
  }
  keys[gid] = key;
}

// ---------------- stage 2: tournament top-512 (exact) ----------------
// Each block sorts a 2048-key chunk descending in LDS, emits its sorted top 512.
__global__ void k_select(const unsigned long long* __restrict__ in,
                         unsigned long long* __restrict__ outk,
                         int in_per_img, int chunks) {
  __shared__ unsigned long long s[2048];
  int b = blockIdx.x / chunks, c = blockIdx.x % chunks;
  int t = threadIdx.x;
  const unsigned long long* base = in + (size_t)b * in_per_img;
  int e0 = c * 2048 + t, e1 = e0 + 1024;
  s[t]        = (e0 < in_per_img) ? base[e0] : 0ull;
  s[t + 1024] = (e1 < in_per_img) ? base[e1] : 0ull;
  for (int k = 2; k <= 2048; k <<= 1) {
    for (int j = k >> 1; j >= 1; j >>= 1) {
      __syncthreads();
      for (int idx = t; idx < 2048; idx += 1024) {
        int ixj = idx ^ j;
        if (ixj > idx) {
          unsigned long long a = s[idx], bb = s[ixj];
          bool up = ((idx & k) == 0);                 // descending order
          if (up ? (a < bb) : (a > bb)) { s[idx] = bb; s[ixj] = a; }
        }
      }
    }
  }
  __syncthreads();
  if (t < 512) outk[((size_t)b * chunks + c) * 512 + t] = s[t];
}

// ---------------- stage 3: gather candidate boxes/scores ----------------
__global__ void k_gather(const float* __restrict__ pred,
                         const unsigned long long* __restrict__ keys512,
                         float4* __restrict__ boxOff, float4* __restrict__ boxPl,
                         float* __restrict__ cscore) {
  int gid = blockIdx.x * blockDim.x + threadIdx.x;
  if (gid >= B * MAXC) return;
  int b = gid / MAXC;
  unsigned long long key = keys512[gid];
  unsigned idx = 0xFFFFFFFFu - (unsigned)(key & 0xFFFFFFFFull);
  float4 xo = make_float4(0, 0, 0, 0), xp = make_float4(0, 0, 0, 0);
  float score = -1.0f;
  if (idx < (unsigned)NPRED) {
    const float* p = pred + ((size_t)b * NPRED + idx) * PREDC;
    float cx = p[0], cy = p[1], w = p[2], h = p[3], obj = p[4];
    float best = -1e30f; int bc = 0;
    for (int c = 0; c < NCLS; ++c) {
      float v = p[5 + c] * obj;
      if (v > best) { best = v; bc = c; }
    }
    score = (best > CONF_THRES) ? best : -1.0f;
    xp = make_float4(cx - 0.5f * w, cy - 0.5f * h, cx + 0.5f * w, cy + 0.5f * h);
    float off = (float)bc * MAX_WH;
    xo = make_float4(xp.x + off, xp.y + off, xp.z + off, xp.w + off);
  }
  boxOff[gid] = xo; boxPl[gid] = xp; cscore[gid] = score;
}

// ---------------- stage 4: greedy NMS + compaction ----------------
__global__ void __launch_bounds__(MAXC)
k_nms(const float4* __restrict__ boxOff, const float4* __restrict__ boxPl,
      const float* __restrict__ cscore,
      float4* __restrict__ fbox, int* __restrict__ fmask) {
  __shared__ float4 sb[MAXC];
  __shared__ float  sa[MAXC];
  __shared__ int    keep[MAXC];
  int b = blockIdx.x, t = threadIdx.x;
  float4 bt = boxOff[b * MAXC + t];
  sb[t] = bt;
  sa[t] = (bt.z - bt.x) * (bt.w - bt.y);
  keep[t] = cscore[b * MAXC + t] > 0.0f;
  for (int i = 0; i < MAXC; ++i) {
    __syncthreads();
    if (keep[i] && t > i) {
      float4 bi = sb[i];
      float lx = fmaxf(bi.x, bt.x), ly = fmaxf(bi.y, bt.y);
      float rx = fminf(bi.z, bt.z), ry = fminf(bi.w, bt.w);
      float iw = fmaxf(rx - lx, 0.0f), ih = fmaxf(ry - ly, 0.0f);
      float inter = iw * ih;
      float iou = inter / (sa[i] + sa[t] - inter + 1e-9f);
      if (iou > IOU_THRES) keep[t] = 0;
    }
  }
  __syncthreads();
  if (t == 0) {   // scores already descending -> compaction preserves rank order
    int rank = 0;
    for (int j = 0; j < MAXC && rank < LIMIT; ++j) {
      if (keep[j]) {
        fbox[b * LIMIT + rank] = boxPl[b * MAXC + j];
        fmask[b * LIMIT + rank] = 1;
        ++rank;
      }
    }
    for (; rank < LIMIT; ++rank) {
      fbox[b * LIMIT + rank] = make_float4(0, 0, 0, 0);
      fmask[b * LIMIT + rank] = 0;
    }
  }
}

// ---------------- zero fill (f16 buffer as u32) ----------------
__global__ void k_zero32(unsigned* __restrict__ p, size_t n) {
  size_t i = (size_t)blockIdx.x * blockDim.x + threadIdx.x;
  if (i < n) p[i] = 0u;
}

// ---------------- weight prep: f32 -> f16 transposed ----------------
__global__ void k_prep(const float* __restrict__ W1, const float* __restrict__ W2,
                       _Float16* __restrict__ W1T, _Float16* __restrict__ W2T) {
  int gid = blockIdx.x * blockDim.x + threadIdx.x;
  if (gid < K1 * NH) {
    int k = gid / NH, n = gid % NH;
    W1T[(size_t)n * K1 + k] = (_Float16)W1[gid];
  } else if (gid < K1 * NH + NH * NH) {
    int g = gid - K1 * NH;
    int k = g / NH, n = g % NH;
    W2T[n * NH + k] = (_Float16)W2[g];
  }
}

// ---------------- NCHW -> NHWC transpose (LDS tiled, both sides coalesced) ----
__global__ void k_nhwc(const float* __restrict__ in, float* __restrict__ out,
                       int C, int H, int W) {
  __shared__ float tile[32][33];
  int x0 = blockIdx.x * 32;
  int c0 = blockIdx.y * 32;
  int bz = blockIdx.z;              // b*H + y
  int b = bz / H, y = bz % H;
  // read: consecutive tx -> consecutive x (coalesced)
  for (int i = threadIdx.y; i < 32; i += 8) {
    int c = c0 + i, x = x0 + threadIdx.x;
    float v = 0.0f;
    if (c < C && x < W) v = in[(((size_t)b * C + c) * H + y) * W + x];
    tile[i][threadIdx.x] = v;
  }
  __syncthreads();
  // write: consecutive tx -> consecutive c (coalesced)
  for (int i = threadIdx.y; i < 32; i += 8) {
    int x = x0 + i, c = c0 + threadIdx.x;
    if (c < C && x < W)
      out[(((size_t)b * H + y) * W + x) * C + c] = tile[threadIdx.x][i];
  }
}

// ---------------- stage 5a: ROI-align, NHWC (coalesced) ----------------
__device__ __forceinline__ void roi_level_nhwc(const float* __restrict__ f,
                                               int H, int W, int C, int cb,
                                               float scale, float4 box,
                                               _Float16* __restrict__ Rrow) {
  float bx = box.x * scale, by = box.y * scale;
  float bw = fmaxf(box.z * scale - bx, 1.0f);
  float bh = fmaxf(box.w * scale - by, 1.0f);
  // precompute 4 sample points (uniform across channels)
  int base00[4], base01[4], base10[4], base11[4];
  float w00[4], w01[4], w10[4], w11[4];
  #pragma unroll
  for (int p = 0; p < 4; ++p) {
    float gx = (p & 1) ? 0.75f : 0.25f;
    float gy = (p >> 1) ? 0.75f : 0.25f;
    float X = fminf(fmaxf(bx + gx * bw, 0.0f), (float)(W - 1));
    float Y = fminf(fmaxf(by + gy * bh, 0.0f), (float)(H - 1));
    int x0 = (int)floorf(X), y0 = (int)floorf(Y);
    int x1i = min(x0 + 1, W - 1), y1i = min(y0 + 1, H - 1);
    float wx = X - (float)x0, wy = Y - (float)y0;
    base00[p] = (y0 * W + x0) * C;   w00[p] = (1.0f - wy) * (1.0f - wx);
    base01[p] = (y0 * W + x1i) * C;  w01[p] = (1.0f - wy) * wx;
    base10[p] = (y1i * W + x0) * C;  w10[p] = wy * (1.0f - wx);
    base11[p] = (y1i * W + x1i) * C; w11[p] = wy * wx;
  }
  for (int c = threadIdx.x; c < C; c += 256) {
    float acc = 0.0f;
    #pragma unroll
    for (int p = 0; p < 4; ++p) {
      acc += f[base00[p] + c] * w00[p] + f[base01[p] + c] * w01[p]
           + f[base10[p] + c] * w10[p] + f[base11[p] + c] * w11[p];
    }
    Rrow[cb + c] = (_Float16)(acc * 0.25f);
  }
}

__global__ void __launch_bounds__(256)
k_roi_nhwc(const float* __restrict__ t1, const float* __restrict__ t2,
           const float* __restrict__ t3, const float* __restrict__ t4,
           const float4* __restrict__ fbox, const int* __restrict__ fmask,
           _Float16* __restrict__ R) {
  int b = blockIdx.x / LIMIT, slot = blockIdx.x % LIMIT;
  if (!fmask[b * LIMIT + slot]) return;        // R pre-zeroed
  float4 box = fbox[b * LIMIT + slot];
  _Float16* Rrow = R + ((size_t)b * MROWS + slot) * K1;
  roi_level_nhwc(t1 + (size_t)b * 128 * 128 * 128, 128, 128, 128,  0,   1.0f / 8.0f,  box, Rrow);
  roi_level_nhwc(t2 + (size_t)b * 64 * 64 * 256,   64,  64,  256,  128, 1.0f / 16.0f, box, Rrow);
  roi_level_nhwc(t3 + (size_t)b * 32 * 32 * 512,   32,  32,  512,  384, 1.0f / 32.0f, box, Rrow);
  roi_level_nhwc(t4 + (size_t)b * 16 * 16 * 1024,  16,  16,  1024, 896, 1.0f / 64.0f, box, Rrow);
}

// ---------------- stage 5b: ROI-align, NCHW fallback (if ws too small) ------
__global__ void k_roi(const float* __restrict__ x1, const float* __restrict__ x2,
                      const float* __restrict__ x3, const float* __restrict__ x4,
                      const float4* __restrict__ fbox, const int* __restrict__ fmask,
                      _Float16* __restrict__ R) {
  int b = blockIdx.x / LIMIT, slot = blockIdx.x % LIMIT;
  if (!fmask[b * LIMIT + slot]) return;
  float4 box = fbox[b * LIMIT + slot];
  const float* f1 = x1 + (size_t)b * 128 * 128 * 128;
  const float* f2 = x2 + (size_t)b * 256 * 64 * 64;
  const float* f3 = x3 + (size_t)b * 512 * 32 * 32;
  const float* f4 = x4 + (size_t)b * 1024 * 16 * 16;
  for (int c = threadIdx.x; c < K1; c += blockDim.x) {
    const float* f; int H; float scale; int cb;
    if (c < 128)      { f = f1; H = 128; scale = 1.0f / 8.0f;  cb = 0;   }
    else if (c < 384) { f = f2; H = 64;  scale = 1.0f / 16.0f; cb = 128; }
    else if (c < 896) { f = f3; H = 32;  scale = 1.0f / 32.0f; cb = 384; }
    else              { f = f4; H = 16;  scale = 1.0f / 64.0f; cb = 896; }
    int W = H;
    const float* fc = f + (size_t)(c - cb) * H * W;
    float bx = box.x * scale, by = box.y * scale;
    float bw = fmaxf(box.z * scale - bx, 1.0f);
    float bh = fmaxf(box.w * scale - by, 1.0f);
    float acc = 0.0f;
    #pragma unroll
    for (int p = 0; p < 4; ++p) {
      float gx = (p & 1) ? 0.75f : 0.25f;
      float gy = (p >> 1) ? 0.75f : 0.25f;
      float X = fminf(fmaxf(bx + gx * bw, 0.0f), (float)(W - 1));
      float Y = fminf(fmaxf(by + gy * bh, 0.0f), (float)(H - 1));
      int x0 = (int)floorf(X), y0 = (int)floorf(Y);
      int x1i = min(x0 + 1, W - 1), y1i = min(y0 + 1, H - 1);
      float wx = X - (float)x0, wy = Y - (float)y0;
      acc += fc[y0 * W + x0]   * (1.0f - wy) * (1.0f - wx)
           + fc[y0 * W + x1i]  * (1.0f - wy) * wx
           + fc[y1i * W + x0]  * wy * (1.0f - wx)
           + fc[y1i * W + x1i] * wy * wx;
    }
    R[((size_t)b * MROWS + slot) * K1 + c] = (_Float16)(acc * 0.25f);
  }
}

// ---------------- WMMA helpers ----------------
// A 16x32 f16 lane layout: lane<16 -> row=lane, K {0..7,16..23}; lane>=16 -> K {8..15,24..31}
__device__ __forceinline__ v16h load_afrag(const _Float16* Arow, int kt, int hs) {
  int ka = kt * 32 + hs * 8;
  v8h alo = *(const v8h*)(Arow + ka);
  v8h ahi = *(const v8h*)(Arow + ka + 16);
  v16h a;
  #pragma unroll
  for (int j = 0; j < 8; ++j) { a[j] = alo[j]; a[j + 8] = ahi[j]; }
  return a;
}

__device__ __forceinline__ void load_bfrag4(v16h* dst, const _Float16* Wt,
                                            size_t ldb, int kt, int hs, int lr) {
  int kb = kt * 32 + hs * 8;
  #pragma unroll
  for (int nt = 0; nt < 4; ++nt) {
    const _Float16* p = Wt + (size_t)(nt * 16 + lr) * ldb + kb;
    v8h lo = *(const v8h*)p;
    v8h hi = *(const v8h*)(p + 16);
    #pragma unroll
    for (int j = 0; j < 8; ++j) { dst[nt][j] = lo[j]; dst[nt][j + 8] = hi[j]; }
  }
}

__device__ __forceinline__ void wmma4(v8f* acc, const v16h a, const v16h* bf) {
  #pragma unroll
  for (int nt = 0; nt < 4; ++nt)
    acc[nt] = __builtin_amdgcn_wmma_f32_16x16x32_f16(
        false, a, false, bf[nt], (short)0, acc[nt], false, false);
}

// ---------------- stage 6a: GEMM1 (MROWS x 1920) @ (1920 x 64) ----------------
// Ping-pong double buffered over k-step pairs: each buffer set is written by
// its loads and consumed directly by its WMMAs -- no loop-carried copies, and
// the kt+1 load clause is in flight while the kt WMMAs run on the matrix pipe.
__global__ void __launch_bounds__(32)
k_gemm1(const _Float16* __restrict__ R, const _Float16* __restrict__ W1T,
        const float* __restrict__ b1, _Float16* __restrict__ H1) {
  int mtile = blockIdx.x, b = blockIdx.y;
  int lane = threadIdx.x, hs = lane >> 4, lr = lane & 15;
  const _Float16* Arow = R + ((size_t)b * MROWS + mtile * 16 + lr) * K1;
  constexpr int KT = K1 / 32;          // 60, even
  v8f acc[4] = {};
  v16h a0, a1, b0[4], b1f[4];
  a0 = load_afrag(Arow, 0, hs);
  load_bfrag4(b0, W1T, K1, 0, hs, lr);
  for (int kt = 0; kt < KT; kt += 2) {
    a1 = load_afrag(Arow, kt + 1, hs);
    load_bfrag4(b1f, W1T, K1, kt + 1, hs, lr);
    wmma4(acc, a0, b0);                 // compute kt, kt+1 loads in flight
    if (kt + 2 < KT) {
      a0 = load_afrag(Arow, kt + 2, hs);
      load_bfrag4(b0, W1T, K1, kt + 2, hs, lr);
    }
    wmma4(acc, a1, b1f);                // compute kt+1, kt+2 loads in flight
  }
  #pragma unroll
  for (int nt = 0; nt < 4; ++nt) {
    #pragma unroll
    for (int r = 0; r < 8; ++r) {
      int row = mtile * 16 + (hs ? 8 + r : r);
      int col = nt * 16 + lr;
      float v = acc[nt][r] + b1[col];
      v = v > 0.0f ? v : NEG_SLOPE * v;
      H1[((size_t)b * MROWS + row) * NH + col] = (_Float16)v;
    }
  }
}

// ---------------- stage 6b: GEMM2 (MROWS x 64) @ (64 x 64) + output ----------------
__global__ void __launch_bounds__(32)
k_gemm2(const _Float16* __restrict__ H1, const _Float16* __restrict__ W2T,
        const float* __restrict__ b2, const int* __restrict__ fmask,
        float* __restrict__ feats) {
  int mtile = blockIdx.x, b = blockIdx.y;
  int lane = threadIdx.x, hs = lane >> 4, lr = lane & 15;
  const _Float16* Arow = H1 + ((size_t)b * MROWS + mtile * 16 + lr) * NH;
  v8f acc[4] = {};
  #pragma unroll
  for (int kt = 0; kt < 2; ++kt) {
    v16h a = load_afrag(Arow, kt, hs);
    v16h bf[4];
    load_bfrag4(bf, W2T, NH, kt, hs, lr);
    wmma4(acc, a, bf);
  }
  #pragma unroll
  for (int nt = 0; nt < 4; ++nt) {
    #pragma unroll
    for (int r = 0; r < 8; ++r) {
      int row = mtile * 16 + (hs ? 8 + r : r);
      if (row < LIMIT) {
        int col = nt * 16 + lr;
        float v = acc[nt][r] + b2[col];
        v = v > 0.0f ? v : NEG_SLOPE * v;
        float m = fmask[b * LIMIT + row] ? 1.0f : 0.0f;
        feats[((size_t)b * LIMIT + row) * 68 + 4 + col] = v * m;
      }
    }
  }
}

// ---------------- stage 7: box/whwh columns + mask output ----------------
__global__ void k_boxout(const float4* __restrict__ fbox, const int* __restrict__ fmask,
                         float* __restrict__ feats, float* __restrict__ maskout) {
  int gid = blockIdx.x * blockDim.x + threadIdx.x;
  if (gid >= B * LIMIT) return;
  float m = fmask[gid] ? 1.0f : 0.0f;
  float4 bx = fbox[gid];
  float inv = 1.0f / (float)IMG;
  float* o = feats + (size_t)gid * 68;
  o[0] = bx.x * inv * m; o[1] = bx.y * inv * m;
  o[2] = bx.z * inv * m; o[3] = bx.w * inv * m;
  maskout[gid] = m;
}

// ---------------- launch ----------------
extern "C" void kernel_launch(void* const* d_in, const int* in_sizes, int n_in,
                              void* d_out, int out_size, void* d_ws, size_t ws_size,
                              hipStream_t stream) {
  const float* pred = (const float*)d_in[0];
  const float* x1   = (const float*)d_in[1];
  const float* x2   = (const float*)d_in[2];
  const float* x3   = (const float*)d_in[3];
  const float* x4   = (const float*)d_in[4];
  const float* W1   = (const float*)d_in[5];
  const float* b1   = (const float*)d_in[6];
  const float* W2   = (const float*)d_in[7];
  const float* b2   = (const float*)d_in[8];

  uint8_t* ws = (uint8_t*)d_ws;
  size_t off = 0;
  auto alloc = [&](size_t bytes) -> void* {
    off = (off + 255) & ~(size_t)255;
    void* p = ws + off;
    off += bytes;
    return p;
  };
  unsigned long long* keysA = (unsigned long long*)alloc((size_t)B * NPAD  * 8);
  unsigned long long* keysB = (unsigned long long*)alloc((size_t)B * 16384 * 8);
  unsigned long long* keysC = (unsigned long long*)alloc((size_t)B * 4096  * 8);
  unsigned long long* keysD = (unsigned long long*)alloc((size_t)B * 1024  * 8);
  unsigned long long* keysE = (unsigned long long*)alloc((size_t)B * 512   * 8);
  float4*   boxOff = (float4*)alloc((size_t)B * MAXC * sizeof(float4));
  float4*   boxPl  = (float4*)alloc((size_t)B * MAXC * sizeof(float4));
  float*    cscore = (float*)alloc((size_t)B * MAXC * 4);
  float4*   fbox   = (float4*)alloc((size_t)B * LIMIT * sizeof(float4));
  int*      fmask  = (int*)alloc((size_t)B * LIMIT * 4);
  _Float16* Rm     = (_Float16*)alloc((size_t)B * MROWS * K1 * 2);
  _Float16* H1m    = (_Float16*)alloc((size_t)B * MROWS * NH * 2);
  _Float16* W1T    = (_Float16*)alloc((size_t)K1 * NH * 2);
  _Float16* W2T    = (_Float16*)alloc((size_t)NH * NH * 2);

  // Optional NHWC scratch for coalesced ROI gathers (~126 MB); gated on ws_size.
  size_t e1 = (size_t)B * 128 * 128 * 128;
  size_t e2 = (size_t)B * 256 * 64 * 64;
  size_t e3 = (size_t)B * 512 * 32 * 32;
  size_t e4 = (size_t)B * 1024 * 16 * 16;
  size_t nhwcNeed = off + 4 * 256 + (e1 + e2 + e3 + e4) * 4;
  bool useNHWC = (ws_size >= nhwcNeed);
  float *t1 = nullptr, *t2 = nullptr, *t3 = nullptr, *t4 = nullptr;
  if (useNHWC) {
    t1 = (float*)alloc(e1 * 4);
    t2 = (float*)alloc(e2 * 4);
    t3 = (float*)alloc(e3 * 4);
    t4 = (float*)alloc(e4 * 4);
  }

  float* feats   = (float*)d_out;
  float* maskout = feats + (size_t)B * LIMIT * 68;

  // 1. keys
  k_keys<<<(B * NPAD + 255) / 256, 256, 0, stream>>>(pred, keysA);
  // 2. tournament top-512 (exact, sorted descending)
  k_select<<<B * 32, 1024, 0, stream>>>(keysA, keysB, NPAD,  32);
  k_select<<<B * 8,  1024, 0, stream>>>(keysB, keysC, 16384, 8);
  k_select<<<B * 2,  1024, 0, stream>>>(keysC, keysD, 4096,  2);
  k_select<<<B * 1,  1024, 0, stream>>>(keysD, keysE, 1024,  1);
  // 3. gather candidates
  k_gather<<<(B * MAXC + 255) / 256, 256, 0, stream>>>(pred, keysE, boxOff, boxPl, cscore);
  // 4. NMS + compaction
  k_nms<<<B, MAXC, 0, stream>>>(boxOff, boxPl, cscore, fbox, fmask);
  // 5. zero R, prep weights, (transpose features,) ROI align
  size_t rwords = (size_t)B * MROWS * K1 / 2;
  k_zero32<<<(unsigned)((rwords + 255) / 256), 256, 0, stream>>>((unsigned*)Rm, rwords);
  k_prep<<<(K1 * NH + NH * NH + 255) / 256, 256, 0, stream>>>(W1, W2, W1T, W2T);
  if (useNHWC) {
    dim3 blk(32, 8);
    k_nhwc<<<dim3(4, 4,  B * 128), blk, 0, stream>>>(x1, t1, 128,  128, 128);
    k_nhwc<<<dim3(2, 8,  B * 64),  blk, 0, stream>>>(x2, t2, 256,  64,  64);
    k_nhwc<<<dim3(1, 16, B * 32),  blk, 0, stream>>>(x3, t3, 512,  32,  32);
    k_nhwc<<<dim3(1, 32, B * 16),  blk, 0, stream>>>(x4, t4, 1024, 16,  16);
    k_roi_nhwc<<<B * LIMIT, 256, 0, stream>>>(t1, t2, t3, t4, fbox, fmask, Rm);
  } else {
    k_roi<<<B * LIMIT, 256, 0, stream>>>(x1, x2, x3, x4, fbox, fmask, Rm);
  }
  // 6. WMMA MLP
  k_gemm1<<<dim3(MROWS / 16, B), 32, 0, stream>>>(Rm, W1T, b1, H1m);
  k_gemm2<<<dim3(MROWS / 16, B), 32, 0, stream>>>(H1m, W2T, b2, fmask, feats);
  // 7. box columns + mask
  k_boxout<<<(B * LIMIT + 255) / 256, 256, 0, stream>>>(fbox, fmask, feats, maskout);
}
